// ODConv_36850819399931
// MI455X (gfx1250) — matrix operand order repaired and checked by reading, not compiled, not run on hardware
//
#include <hip/hip_runtime.h>
#include <cstdint>
#include <cstddef>

typedef _Float16 v8h  __attribute__((ext_vector_type(8)));
typedef _Float16 v16h __attribute__((ext_vector_type(16)));
typedef float    v8f  __attribute__((ext_vector_type(8)));

#define B_    16
#define CIN_  256
#define COUT_ 256
#define KK_   7
#define L_    4096
#define T_    (L_ + KK_ - 1)   /* 4102 */
#define RED_  64
#define XROWS 4368             /* 8 zero pad rows + 4096 data + 264 zero pad rows */
#define XPAD  8                /* row p holds x[:, p-8, :]; p<8 and p>=4104 zero */
#define TWG_  256              /* conv t tile per workgroup */
#define COWG_ 64               /* conv co tile per workgroup */

// ---------------------------------------------------------------------------
// Kernel 0a: zero the guard rows of the transposed f16 x buffer.
// ---------------------------------------------------------------------------
__global__ __launch_bounds__(128) void padzero_kernel(_Float16* __restrict__ xt) {
  const int row = (blockIdx.x < XPAD) ? blockIdx.x : (L_ + blockIdx.x);
  uint32_t* p = (uint32_t*)(xt + ((size_t)blockIdx.y * XROWS + row) * CIN_);
  p[threadIdx.x] = 0u;
}

// ---------------------------------------------------------------------------
// Kernel 0b: transpose + convert: xt[b][t+XPAD][ci] = (f16) x[b][ci][t]
// ---------------------------------------------------------------------------
__global__ __launch_bounds__(256) void transpose_kernel(const float* __restrict__ x,
                                                        _Float16* __restrict__ xt) {
  __shared__ _Float16 tile[64][65];
  const int t0 = blockIdx.x * 64;
  const int c0 = blockIdx.y * 64;
  const int b  = blockIdx.z;
  const int c  = threadIdx.x & 63;
  const int r  = threadIdx.x >> 6;  // 0..3
#pragma unroll
  for (int rr = 0; rr < 16; ++rr) {
    const int ci = rr * 4 + r;
    tile[ci][c] = (_Float16)x[((size_t)(b * CIN_ + c0 + ci)) * L_ + t0 + c];
  }
  __syncthreads();
#pragma unroll
  for (int ww = 0; ww < 16; ++ww) {
    const int tt = ww * 4 + r;
    xt[((size_t)b * XROWS + (t0 + tt + XPAD)) * CIN_ + c0 + c] = tile[c][tt];
  }
}

// ---------------------------------------------------------------------------
// Kernel 1: AdaptiveAvgPool1d(1)
// ---------------------------------------------------------------------------
__global__ __launch_bounds__(256) void pool_kernel(const float* __restrict__ x,
                                                   float* __restrict__ pooled) {
  const int row = blockIdx.x;  // b*CIN + ci
  const float4* xr = (const float4*)(x + (size_t)row * L_);
  float s = 0.f;
  for (int i = threadIdx.x; i < L_ / 4; i += 256) {
    float4 v = xr[i];
    s += v.x + v.y + v.z + v.w;
  }
  __shared__ float red[256];
  red[threadIdx.x] = s;
  __syncthreads();
  for (int off = 128; off > 0; off >>= 1) {
    if (threadIdx.x < off) red[threadIdx.x] += red[threadIdx.x + off];
    __syncthreads();
  }
  if (threadIdx.x == 0) pooled[row] = red[0] * (1.f / (float)L_);
}

// ---------------------------------------------------------------------------
// Kernel 2: SE branch
// ---------------------------------------------------------------------------
__global__ __launch_bounds__(256) void att_kernel(const float* __restrict__ pooled,
                                                  const float* __restrict__ w1,
                                                  const float* __restrict__ b1,
                                                  const float* __restrict__ w2,
                                                  const float* __restrict__ b2,
                                                  float* __restrict__ out, int outdim) {
  __shared__ float h[B_ * RED_];
  const int tid = threadIdx.x;
  for (int idx = tid; idx < B_ * RED_; idx += 256) {
    int bb = idx >> 6, r = idx & (RED_ - 1);
    const float* p = pooled + bb * CIN_;
    const float* w = w1 + r * CIN_;
    float acc = b1[r];
    for (int c = 0; c < CIN_; ++c) acc = fmaf(p[c], w[c], acc);
    h[idx] = fmaxf(acc, 0.f);
  }
  __syncthreads();
  const int total = B_ * outdim;
  for (int idx = tid; idx < total; idx += 256) {
    int bb = idx / outdim, o = idx - bb * outdim;
    const float* hh = h + bb * RED_;
    const float* w = w2 + o * RED_;
    float acc = b2[o];
    for (int r = 0; r < RED_; ++r) acc = fmaf(hh[r], w[r], acc);
    out[idx] = 1.f / (1.f + __expf(-acc));
  }
}

// ---------------------------------------------------------------------------
// Kernel 3: dynamic transposed conv, WMMA f16 + async global->LDS x staging.
// ---------------------------------------------------------------------------
__global__ __launch_bounds__(256) void odconv_main(
    const _Float16* __restrict__ xt, const float* __restrict__ weight,
    const float* __restrict__ bias,
    const float* __restrict__ katt, const float* __restrict__ satt,
    const float* __restrict__ catt, const float* __restrict__ oatt,
    float* __restrict__ out) {
  constexpr int CI_CH = 32;    // ci chunk == WMMA K
  constexpr int CIP   = 40;    // padded ci stride (halves)
  constexpr int XT    = 264;   // staged t rows

  __shared__ __attribute__((aligned(16))) _Float16 xsh[XT * CIP];
  __shared__ __attribute__((aligned(16))) _Float16 wsh[KK_ * COWG_ * CIP];

  const int b    = blockIdx.z;
  const int co0  = blockIdx.y * COWG_;
  const int t0   = blockIdx.x * TWG_;
  const int tid  = threadIdx.x;
  const int lane = tid & 31;
  const int wave = tid >> 5;
  const int half = lane >> 4;
  const int l16  = lane & 15;
  const int wbco = (wave & 1) * 32;
  const int wbt  = (wave >> 1) * 64;

  const uint32_t xsh_base = (uint32_t)(uintptr_t)&xsh[0];
  const _Float16* xbase = xt + (size_t)b * XROWS * CIN_;

  float ka[KK_];
#pragma unroll
  for (int k = 0; k < KK_; ++k) ka[k] = katt[b * KK_ + k];

  const v8f zero8 = {0.f, 0.f, 0.f, 0.f, 0.f, 0.f, 0.f, 0.f};
  v8f acc[2][4];
#pragma unroll
  for (int i = 0; i < 2; ++i)
#pragma unroll
    for (int j = 0; j < 4; ++j) acc[i][j] = zero8;

  for (int ci0 = 0; ci0 < CIN_; ci0 += CI_CH) {
    __syncthreads();

    // async DMA of the shifted x window (hits zero guard rows when OOB)
#pragma unroll
    for (int it = 0; it < 5; ++it) {
      const int g = tid + it * 256;
      if (g < XT * 4) {
        const int tt = g >> 2, q = g & 3;
        const uint32_t ldsoff = xsh_base + (uint32_t)(tt * (CIP * 2) + q * 16);
        const uint32_t voff =
            (uint32_t)((t0 + 2 + tt) * (CIN_ * 2) + (ci0 * 2 + q * 16));
        asm volatile("global_load_async_to_lds_b128 %0, %1, %2"
                     :: "v"(ldsoff), "v"(voff), "s"(xbase)
                     : "memory");
      }
    }

    // weights with k_att and c_att folded in
    for (int pr = tid; pr < COWG_ * CI_CH; pr += 256) {
      const int co = pr >> 5, ci = pr & 31;
      const float ca = catt[b * CIN_ + ci0 + ci];
      const float* wp = weight + ((size_t)(co0 + co) * CIN_ + (ci0 + ci)) * KK_;
#pragma unroll
      for (int k = 0; k < KK_; ++k)
        wsh[(k * COWG_ + co) * CIP + ci] = (_Float16)(wp[k] * (ka[k] * ca));
    }

    asm volatile("s_wait_asynccnt 0x0" ::: "memory");
    __syncthreads();

    for (int k = 0; k < KK_; ++k) {
      v16h afrag[2];
#pragma unroll
      for (int cf = 0; cf < 2; ++cf) {
        const _Float16* ap =
            &wsh[(k * COWG_ + wbco + cf * 16 + l16) * CIP + half * 8];
        v8h a0 = *(const v8h*)ap;
        v8h a1 = *(const v8h*)(ap + 16);
        afrag[cf] = __builtin_shufflevector(a0, a1, 0, 1, 2, 3, 4, 5, 6, 7, 8, 9,
                                            10, 11, 12, 13, 14, 15);
      }
      v16h bfrag[4];
#pragma unroll
      for (int nt = 0; nt < 4; ++nt) {
        const int tloc = wbt + nt * 16 + l16 + (KK_ - 1) - k;
        const _Float16* bp = &xsh[tloc * CIP + half * 8];
        v8h b0 = *(const v8h*)bp;
        v8h b1 = *(const v8h*)(bp + 16);
        bfrag[nt] = __builtin_shufflevector(b0, b1, 0, 1, 2, 3, 4, 5, 6, 7, 8, 9,
                                            10, 11, 12, 13, 14, 15);
      }
#pragma unroll
      for (int nt = 0; nt < 4; ++nt) {
        acc[0][nt] = __builtin_amdgcn_wmma_f32_16x16x32_f16(
            false, afrag[0], false, bfrag[nt], (short)0, acc[0][nt], false, false);
        acc[1][nt] = __builtin_amdgcn_wmma_f32_16x16x32_f16(
            false, afrag[1], false, bfrag[nt], (short)0, acc[1][nt], false, false);
      }
    }
  }

  const float sb = satt[b];
#pragma unroll
  for (int cf = 0; cf < 2; ++cf) {
#pragma unroll
    for (int nt = 0; nt < 4; ++nt) {
      const int t = t0 + wbt + nt * 16 + l16;
      if (t < T_) {
#pragma unroll
        for (int r = 0; r < 8; ++r) {
          const int co = co0 + wbco + cf * 16 + half * 8 + r;
          const float v =
              acc[cf][nt][r] * (sb * oatt[b * COUT_ + co]) + bias[co];
          out[((size_t)(b * COUT_ + co)) * T_ + t] = v;
        }
      }
    }
  }
}

extern "C" void kernel_launch(void* const* d_in, const int* in_sizes, int n_in,
                              void* d_out, int out_size, void* d_ws, size_t ws_size,
                              hipStream_t stream) {
  const float* x      = (const float*)d_in[0];
  const float* weight = (const float*)d_in[1];
  const float* bias   = (const float*)d_in[2];
  const float* kw1 = (const float*)d_in[3];
  const float* kb1 = (const float*)d_in[4];
  const float* kw2 = (const float*)d_in[5];
  const float* kb2 = (const float*)d_in[6];
  const float* sw1 = (const float*)d_in[7];
  const float* sb1 = (const float*)d_in[8];
  const float* sw2 = (const float*)d_in[9];
  const float* sb2 = (const float*)d_in[10];
  const float* cw1 = (const float*)d_in[11];
  const float* cb1 = (const float*)d_in[12];
  const float* cw2 = (const float*)d_in[13];
  const float* cb2 = (const float*)d_in[14];
  const float* ow1 = (const float*)d_in[15];
  const float* ob1 = (const float*)d_in[16];
  const float* ow2 = (const float*)d_in[17];
  const float* ob2 = (const float*)d_in[18];

  float* ws     = (float*)d_ws;
  float* pooled = ws;
  float* katt   = ws + 4096;
  float* satt   = ws + 4096 + 112;
  float* catt   = ws + 4096 + 128;
  float* oatt   = catt + 4096;
  _Float16* xt  = (_Float16*)((char*)d_ws + 65536);

  padzero_kernel<<<dim3(XROWS - L_, B_), 128, 0, stream>>>(xt);
  transpose_kernel<<<dim3(L_ / 64, CIN_ / 64, B_), 256, 0, stream>>>(x, xt);
  pool_kernel<<<B_ * CIN_, 256, 0, stream>>>(x, pooled);
  att_kernel<<<1, 256, 0, stream>>>(pooled, kw1, kb1, kw2, kb2, katt, KK_);
  att_kernel<<<1, 256, 0, stream>>>(pooled, sw1, sb1, sw2, sb2, satt, 1);
  att_kernel<<<1, 256, 0, stream>>>(pooled, cw1, cb1, cw2, cb2, catt, CIN_);
  att_kernel<<<1, 256, 0, stream>>>(pooled, ow1, ob1, ow2, ob2, oatt, COUT_);

  dim3 grid((T_ + TWG_ - 1) / TWG_, COUT_ / COWG_, B_);
  odconv_main<<<grid, 256, 0, stream>>>(xt, weight, bias, katt, satt, catt, oatt,
                                        (float*)d_out);
}
